// SentinelFlowGATWithEdge_4758823764429
// MI455X (gfx1250) — compile-verified, matrix operand order checked by reading, not run on hardware
//
#include <hip/hip_runtime.h>
#include <hip/hip_bf16.h>

#define NNODES 50000
#define NEDGES 800000
#define IN_DIM 8
#define ED_DIM 3
#define OUT_C  32
#define HEADS  4
#define TOT_C  128

typedef __attribute__((ext_vector_type(16))) _Float16 v16h;
typedef __attribute__((ext_vector_type(8)))  _Float16 v8h;
typedef __attribute__((ext_vector_type(8)))  float    v8f;

// ---------------------------------------------------------------- utilities

__global__ void k_fill(float* __restrict__ p, float v, int n) {
    int i = blockIdx.x * blockDim.x + threadIdx.x;
    if (i < n) p[i] = v;
}

__device__ inline void atomicMaxF(float* addr, float val) {
    // monotone bit trick: works for mixed signs when initialized to -inf
    if (val >= 0.0f) atomicMax((int*)addr, __float_as_int(val));
    else             atomicMin((unsigned int*)addr, __float_as_uint(val));
}

// ------------------------------------------------- layer-0 transform (K=8)

__global__ void k_gemm_small(const float* __restrict__ X, const float* __restrict__ W,
                             const float* __restrict__ b, float* __restrict__ Y,
                             int n, int kin, int kout) {
    int i = blockIdx.x * blockDim.x + threadIdx.x;
    if (i >= n * kout) return;
    int r = i / kout, c = i % kout;
    float acc = b[c];
#pragma unroll 8
    for (int k = 0; k < kin; ++k) acc += X[r * kin + k] * W[k * kout + c];
    Y[i] = acc;
}

// -------------------------------------------------- f16 conversion helpers

__global__ void k_w_to_f16_t(const float* __restrict__ W, _Float16* __restrict__ Wt,
                             int K, int KOUT) {
    int i = blockIdx.x * blockDim.x + threadIdx.x;
    if (i >= K * KOUT) return;
    int k = i / KOUT, c = i % KOUT;
    Wt[c * K + k] = (_Float16)W[i];   // transposed: [KOUT, K]
}

// ------------------------------------------- WMMA GEMM: [N,128]f16 @ [128,KOUT]
// A row-major f16 [N,128]; Wt = W^T row-major f16 [KOUT,128]; Y f32 [N,KOUT]+bias.
// One wave per 16-row stripe: the 4 A fragments (K=128) are loaded once into
// registers and reused across all KOUT/16 column tiles (8x A-traffic reduction
// for KOUT=128). B rows are L2/WGP$-resident weights.

__global__ void k_wmma_gemm(const _Float16* __restrict__ A,
                            const _Float16* __restrict__ Wt,
                            const float* __restrict__ bias,
                            float* __restrict__ Y,
                            int n, int kout) {
    const int K = 128;
    int wave = blockIdx.x * (blockDim.x >> 5) + (threadIdx.x >> 5);
    if (wave >= (n >> 4)) return;        // wave-uniform: EXEC stays all-ones

    int lane = threadIdx.x & 31;
    int m    = lane & 15;
    int kgrp = (lane >> 4) << 3;         // 0 or 8 (ISA 16-bit A-matrix layout)
    const _Float16* Arow = A + (size_t)(wave * 16 + m) * K;

    // Resident A fragments covering K = 0..127
    v16h afrag[4];
#pragma unroll
    for (int t = 0; t < 4; ++t) {
        v8h alo = *(const v8h*)(Arow + t * 32 + kgrp);
        v8h ahi = *(const v8h*)(Arow + t * 32 + 16 + kgrp);
        afrag[t] = __builtin_shufflevector(alo, ahi,
                                           0,1,2,3,4,5,6,7,8,9,10,11,12,13,14,15);
    }

    int ntc = kout >> 4;
    int mbase = wave * 16 + ((lane >> 4) << 3);
    for (int ct = 0; ct < ntc; ++ct) {
        const _Float16* Brow = Wt + (size_t)(ct * 16 + m) * K;
        __builtin_prefetch(Brow + K, 0, 1);   // following Wt row (ct*16+m+1)
        v8f c = {};
#pragma unroll
        for (int t = 0; t < 4; ++t) {
            v8h blo = *(const v8h*)(Brow + t * 32 + kgrp);
            v8h bhi = *(const v8h*)(Brow + t * 32 + 16 + kgrp);
            v16h b = __builtin_shufflevector(blo, bhi,
                                             0,1,2,3,4,5,6,7,8,9,10,11,12,13,14,15);
            c = __builtin_amdgcn_wmma_f32_16x16x32_f16(false, afrag[t], false, b,
                                                       (short)0, c, false, false);
        }
        // C/D layout: VGPR r -> M = r (lanes 0-15) or r+8 (lanes 16-31); N = lane&15
        int nIdx = ct * 16 + (lane & 15);
        float bv = bias[nIdx];
#pragma unroll
        for (int r = 0; r < 8; ++r)
            Y[(size_t)(mbase + r) * kout + nIdx] = c[r] + bv;
    }
}

// -------------------------------------------------------------- edge phase

__global__ void k_edge_score(const float* __restrict__ xl, const float* __restrict__ xr,
                             const float* __restrict__ eattr, const float* __restrict__ We,
                             const float* __restrict__ att,
                             const int* __restrict__ src, const int* __restrict__ dst,
                             float* __restrict__ score, float* __restrict__ mbuf,
                             int ne, int H, int outc, int totc) {
    int t = blockIdx.x * blockDim.x + threadIdx.x;
    if (t >= ne * H) return;
    int e = t / H, h = t % H;
    int s = src[e], d = dst[e];
    float a0 = eattr[e * 3 + 0], a1 = eattr[e * 3 + 1], a2 = eattr[e * 3 + 2];
    const float* xls = xl + (size_t)s * totc + h * outc;
    const float* xrd = xr + (size_t)d * totc + h * outc;
    float acc = 0.0f;
    for (int cidx = 0; cidx < outc; ++cidx) {
        int cc = h * outc + cidx;
        float ee = a0 * We[cc] + a1 * We[totc + cc] + a2 * We[2 * totc + cc];
        float msg = xls[cidx] + xrd[cidx] + ee;
        msg = (msg > 0.0f) ? msg : 0.2f * msg;   // leaky relu
        acc += msg * att[h * outc + cidx];
    }
    score[t] = acc;
    atomicMaxF(&mbuf[d * H + h], acc);
}

__global__ void k_edge_exp(float* __restrict__ score, const float* __restrict__ mbuf,
                           float* __restrict__ sbuf, const int* __restrict__ dst,
                           int ne, int H) {
    int t = blockIdx.x * blockDim.x + threadIdx.x;
    if (t >= ne * H) return;
    int e = t / H, h = t % H;
    int d = dst[e];
    float ex = expf(score[t] - mbuf[d * H + h]);
    score[t] = ex;
    atomicAdd(&sbuf[d * H + h], ex);
}

__global__ void k_edge_agg(const float* __restrict__ xl, const float* __restrict__ score,
                           const float* __restrict__ sbuf,
                           const int* __restrict__ src, const int* __restrict__ dst,
                           float* __restrict__ agg, int ne, int H, int outc, int totc) {
    int t = blockIdx.x * blockDim.x + threadIdx.x;
    if (t >= ne * totc) return;
    int e = t / totc, c = t % totc;
    int h = c / outc;
    int s = src[e], d = dst[e];
    float alpha = score[e * H + h] / (sbuf[d * H + h] + 1e-16f);
    atomicAdd(&agg[(size_t)d * totc + c], alpha * xl[(size_t)s * totc + c]);
}

// --------------------------------------------------------------- batchnorm

__global__ void k_bn_stats(const float* __restrict__ agg, const float* __restrict__ bias,
                           float* __restrict__ stats, int n, int totc) {
    __shared__ float ssum[256], ssq[256];
    int c = blockIdx.x;
    float bv = bias[c];
    float sum = 0.0f, sq = 0.0f;
    for (int i = threadIdx.x; i < n; i += blockDim.x) {
        float v = agg[(size_t)i * totc + c] + bv;
        sum += v; sq += v * v;
    }
    ssum[threadIdx.x] = sum; ssq[threadIdx.x] = sq;
    __syncthreads();
    for (int s = blockDim.x >> 1; s > 0; s >>= 1) {
        if (threadIdx.x < (unsigned)s) {
            ssum[threadIdx.x] += ssum[threadIdx.x + s];
            ssq[threadIdx.x]  += ssq[threadIdx.x + s];
        }
        __syncthreads();
    }
    if (threadIdx.x == 0) {
        float mu = ssum[0] / (float)n;
        stats[c * 2 + 0] = mu;
        stats[c * 2 + 1] = ssq[0] / (float)n - mu * mu;  // biased var (matches jnp.var)
    }
}

__global__ void k_bn_elu(const float* __restrict__ agg, const float* __restrict__ bias,
                         const float* __restrict__ stats,
                         const float* __restrict__ g, const float* __restrict__ be,
                         float* __restrict__ hf, _Float16* __restrict__ hh,
                         int n, int totc) {
    int i = blockIdx.x * blockDim.x + threadIdx.x;
    if (i >= n * totc) return;
    int c = i % totc;
    float v = agg[i] + bias[c];
    float y = (v - stats[c * 2]) * rsqrtf(stats[c * 2 + 1] + 1e-5f) * g[c] + be[c];
    y = (y > 0.0f) ? y : (expf(y) - 1.0f);   // ELU
    hf[i] = y;
    if (hh) hh[i] = (_Float16)y;
}

// ------------------------------------------------------------- classifier

__global__ void k_classifier(const float* __restrict__ h, const float* __restrict__ Wc,
                             const float* __restrict__ bc, float* __restrict__ out, int n) {
    int i = blockIdx.x * blockDim.x + threadIdx.x;
    if (i >= n) return;
    float z0 = bc[0], z1 = bc[1];
    const float* hr = h + (size_t)i * OUT_C;
#pragma unroll
    for (int k = 0; k < OUT_C; ++k) {
        z0 += hr[k] * Wc[k * 2 + 0];
        z1 += hr[k] * Wc[k * 2 + 1];
    }
    float m = fmaxf(z0, z1);
    float lse = m + logf(expf(z0 - m) + expf(z1 - m));
    out[i * 2 + 0] = z0 - lse;
    out[i * 2 + 1] = z1 - lse;
}

// ------------------------------------------------------------------ launch

static inline char* wsAlloc(char*& p, size_t bytes) {
    char* r = p;
    p += (bytes + 255) & ~((size_t)255);
    return r;
}

extern "C" void kernel_launch(void* const* d_in, const int* in_sizes, int n_in,
                              void* d_out, int out_size, void* d_ws, size_t ws_size,
                              hipStream_t stream) {
    (void)in_sizes; (void)n_in; (void)out_size; (void)ws_size;
    const float* x     = (const float*)d_in[0];
    const int*   eidx  = (const int*)  d_in[1];
    const float* eattr = (const float*)d_in[2];
    const float* Wl0 = (const float*)d_in[3],  *bl0 = (const float*)d_in[4];
    const float* Wr0 = (const float*)d_in[5],  *br0 = (const float*)d_in[6];
    const float* We0 = (const float*)d_in[7],  *att0 = (const float*)d_in[8];
    const float* bias0 = (const float*)d_in[9], *g0 = (const float*)d_in[10], *be0 = (const float*)d_in[11];
    const float* Wl1 = (const float*)d_in[12], *bl1 = (const float*)d_in[13];
    const float* Wr1 = (const float*)d_in[14], *br1 = (const float*)d_in[15];
    const float* We1 = (const float*)d_in[16], *att1 = (const float*)d_in[17];
    const float* bias1 = (const float*)d_in[18], *g1 = (const float*)d_in[19], *be1 = (const float*)d_in[20];
    const float* Wl2 = (const float*)d_in[21], *bl2 = (const float*)d_in[22];
    const float* Wr2 = (const float*)d_in[23], *br2 = (const float*)d_in[24];
    const float* We2 = (const float*)d_in[25], *att2 = (const float*)d_in[26];
    const float* bias2 = (const float*)d_in[27], *g2 = (const float*)d_in[28], *be2 = (const float*)d_in[29];
    const float* Wc = (const float*)d_in[30], *bc = (const float*)d_in[31];

    const int* srcI = eidx;
    const int* dstI = eidx + NEDGES;

    char* p = (char*)d_ws;
    float*    xl    = (float*)   wsAlloc(p, (size_t)NNODES * TOT_C * 4);
    float*    xr    = (float*)   wsAlloc(p, (size_t)NNODES * TOT_C * 4);
    float*    agg   = (float*)   wsAlloc(p, (size_t)NNODES * TOT_C * 4);
    float*    hf    = (float*)   wsAlloc(p, (size_t)NNODES * TOT_C * 4);
    _Float16* hh    = (_Float16*)wsAlloc(p, (size_t)NNODES * TOT_C * 2);
    float*    mbuf  = (float*)   wsAlloc(p, (size_t)NNODES * HEADS * 4);
    float*    sbuf  = (float*)   wsAlloc(p, (size_t)NNODES * HEADS * 4);
    float*    score = (float*)   wsAlloc(p, (size_t)NEDGES * HEADS * 4);
    _Float16* wta   = (_Float16*)wsAlloc(p, 128 * 128 * 2);
    _Float16* wtb   = (_Float16*)wsAlloc(p, 128 * 128 * 2);
    float*    stats = (float*)   wsAlloc(p, 256 * 4);

    const float NEG_INF = -__builtin_huge_valf();
    auto g1d = [](int n) { return (n + 255) / 256; };

    auto edge_phase = [&](const float* We, const float* att, const float* bias,
                          const float* g, const float* be, int H, int outc, int totc,
                          float* houtf, _Float16* houth) {
        k_fill<<<g1d(NNODES * H), 256, 0, stream>>>(mbuf, NEG_INF, NNODES * H);
        k_fill<<<g1d(NNODES * H), 256, 0, stream>>>(sbuf, 0.0f, NNODES * H);
        k_fill<<<g1d(NNODES * totc), 256, 0, stream>>>(agg, 0.0f, NNODES * totc);
        k_edge_score<<<g1d(NEDGES * H), 256, 0, stream>>>(
            xl, xr, eattr, We, att, srcI, dstI, score, mbuf, NEDGES, H, outc, totc);
        k_edge_exp<<<g1d(NEDGES * H), 256, 0, stream>>>(score, mbuf, sbuf, dstI, NEDGES, H);
        k_edge_agg<<<g1d(NEDGES * totc), 256, 0, stream>>>(
            xl, score, sbuf, srcI, dstI, agg, NEDGES, H, outc, totc);
        k_bn_stats<<<totc, 256, 0, stream>>>(agg, bias, stats, NNODES, totc);
        k_bn_elu<<<g1d(NNODES * totc), 256, 0, stream>>>(
            agg, bias, stats, g, be, houtf, houth, NNODES, totc);
    };

    // ---- Layer 0 (K=8: VALU transform) ----
    k_gemm_small<<<g1d(NNODES * TOT_C), 256, 0, stream>>>(x, Wl0, bl0, xl, NNODES, IN_DIM, TOT_C);
    k_gemm_small<<<g1d(NNODES * TOT_C), 256, 0, stream>>>(x, Wr0, br0, xr, NNODES, IN_DIM, TOT_C);
    edge_phase(We0, att0, bias0, g0, be0, HEADS, OUT_C, TOT_C, hf, hh);

    // ---- Layer 1 (WMMA transforms, K=128 -> 128) ----
    k_w_to_f16_t<<<g1d(128 * 128), 256, 0, stream>>>(Wl1, wta, 128, 128);
    k_w_to_f16_t<<<g1d(128 * 128), 256, 0, stream>>>(Wr1, wtb, 128, 128);
    {
        int waves = NNODES / 16;             // one wave per 16-row stripe
        int blocks = (waves + 3) / 4;
        k_wmma_gemm<<<blocks, 128, 0, stream>>>(hh, wta, bl1, xl, NNODES, TOT_C);
        k_wmma_gemm<<<blocks, 128, 0, stream>>>(hh, wtb, br1, xr, NNODES, TOT_C);
    }
    edge_phase(We1, att1, bias1, g1, be1, HEADS, OUT_C, TOT_C, hf, hh);

    // ---- Layer 2 (WMMA transforms, K=128 -> 32; heads=1) ----
    k_w_to_f16_t<<<g1d(128 * 32), 256, 0, stream>>>(Wl2, wta, 128, 32);
    k_w_to_f16_t<<<g1d(128 * 32), 256, 0, stream>>>(Wr2, wtb, 128, 32);
    {
        int waves = NNODES / 16;
        int blocks = (waves + 3) / 4;
        k_wmma_gemm<<<blocks, 128, 0, stream>>>(hh, wta, bl2, xl, NNODES, OUT_C);
        k_wmma_gemm<<<blocks, 128, 0, stream>>>(hh, wtb, br2, xr, NNODES, OUT_C);
    }
    edge_phase(We2, att2, bias2, g2, be2, 1, OUT_C, OUT_C, hf, nullptr);

    // ---- Classifier + log_softmax ----
    k_classifier<<<g1d(NNODES), 256, 0, stream>>>(hf, Wc, bc, (float*)d_out, NNODES);
}
